// MMoE_86251533238442
// MI455X (gfx1250) — matrix operand order repaired and checked by reading, not compiled, not run on hardware
//
#include <hip/hip_runtime.h>

typedef _Float16 v16h __attribute__((ext_vector_type(16)));
typedef _Float16 v8h  __attribute__((ext_vector_type(8)));
typedef float    v8f  __attribute__((ext_vector_type(8)));

#define B_  512
#define F_  64
#define S_  400
#define E_  8
#define T_  4
#define C_  16
#define ST_ (S_ / 16)          // 25 s-tiles per batch row
#define NTILE_ (B_ * ST_)      // 12800 wave-tiles

// ---------------------------------------------------------------------------
// Kernel 1: x (B,F,S) f32  ->  xT (B,S,F) f16   (transpose F<->S + convert)
// Reads coalesced across lanes (consecutive s); each lane writes one 128B row.
// ---------------------------------------------------------------------------
__global__ void prep_x(const float* __restrict__ x, _Float16* __restrict__ xT) {
    const int idx = blockIdx.x * blockDim.x + threadIdx.x;   // one per (b,s)
    const int b = idx / S_;
    const int s = idx % S_;
    const float* src = x + (size_t)b * F_ * S_ + s;
    __attribute__((aligned(16))) _Float16 h[F_];
#pragma unroll
    for (int f = 0; f < F_; ++f) h[f] = (_Float16)src[(size_t)f * S_];
    _Float16* dst = xT + ((size_t)b * S_ + s) * F_;
#pragma unroll
    for (int j = 0; j < F_ / 8; ++j) *(v8h*)(dst + j * 8) = *(const v8h*)(h + j * 8);
}

// ---------------------------------------------------------------------------
// Kernel 2: conv_w (E,C,F,3) f32 -> w16 [e][k][c][f] f16 (A-fragment friendly)
// ---------------------------------------------------------------------------
__global__ void prep_w(const float* __restrict__ conv_w, _Float16* __restrict__ w16) {
    for (int idx = blockIdx.x * blockDim.x + threadIdx.x; idx < E_ * 3 * C_ * F_;
         idx += gridDim.x * blockDim.x) {
        const int f = idx % F_;
        const int c = (idx / F_) % C_;
        const int k = (idx / (F_ * C_)) % 3;
        const int e = idx / (F_ * C_ * 3);
        w16[idx] = (_Float16)conv_w[(((size_t)e * C_ + c) * F_ + f) * 3 + k];
    }
}

// ---------------------------------------------------------------------------
// Kernel 3: gate logits + softmax, FP32, deterministic tree reduction.
// One block per b; gk (3.3MB) lives in L2 across all 512 blocks.
// gates out: (T,B,E) f32
// ---------------------------------------------------------------------------
__global__ void gates_kernel(const float* __restrict__ x, const float* __restrict__ gk,
                             const float* __restrict__ gate_bias,
                             float* __restrict__ gates) {
    const int b   = blockIdx.x;
    const int tid = threadIdx.x;
    const int wave = tid >> 5;
    const int lane = tid & 31;

    __shared__ float sparts[8][T_ * E_];
    __shared__ float slog[T_ * E_];

    float acc[T_ * E_];
#pragma unroll
    for (int j = 0; j < T_ * E_; ++j) acc[j] = 0.0f;

    const float* xr = x + (size_t)b * (F_ * S_);
    for (int i = tid; i < F_ * S_; i += 256) {
        const float xv = xr[i];
#pragma unroll
        for (int t = 0; t < T_; ++t) {
            const float4* g4 = (const float4*)(gk + ((size_t)t * (F_ * S_) + i) * E_);
            const float4 g0 = g4[0];
            const float4 g1 = g4[1];
            acc[t * E_ + 0] = fmaf(xv, g0.x, acc[t * E_ + 0]);
            acc[t * E_ + 1] = fmaf(xv, g0.y, acc[t * E_ + 1]);
            acc[t * E_ + 2] = fmaf(xv, g0.z, acc[t * E_ + 2]);
            acc[t * E_ + 3] = fmaf(xv, g0.w, acc[t * E_ + 3]);
            acc[t * E_ + 4] = fmaf(xv, g1.x, acc[t * E_ + 4]);
            acc[t * E_ + 5] = fmaf(xv, g1.y, acc[t * E_ + 5]);
            acc[t * E_ + 6] = fmaf(xv, g1.z, acc[t * E_ + 6]);
            acc[t * E_ + 7] = fmaf(xv, g1.w, acc[t * E_ + 7]);
        }
    }
    // wave32 shuffle reduction, then fixed-order cross-wave sum (deterministic)
#pragma unroll
    for (int j = 0; j < T_ * E_; ++j) {
        float v = acc[j];
        for (int off = 16; off > 0; off >>= 1) v += __shfl_down(v, off, 32);
        if (lane == 0) sparts[wave][j] = v;
    }
    __syncthreads();
    if (tid < T_ * E_) {
        float s = gate_bias[tid];     // (T,1,E) flat == 32 values
#pragma unroll
        for (int w = 0; w < 8; ++w) s += sparts[w][tid];
        slog[tid] = s;
    }
    __syncthreads();
    if (tid < T_) {
        float l[E_];
        float m = -1e30f;
#pragma unroll
        for (int e = 0; e < E_; ++e) { l[e] = slog[tid * E_ + e]; m = fmaxf(m, l[e]); }
        float ssum = 0.0f;
#pragma unroll
        for (int e = 0; e < E_; ++e) { l[e] = __expf(l[e] - m); ssum += l[e]; }
        const float inv = 1.0f / ssum;
#pragma unroll
        for (int e = 0; e < E_; ++e)
            gates[((size_t)tid * B_ + b) * E_ + e] = l[e] * inv;
    }
}

// ---------------------------------------------------------------------------
// Kernel 4: fused conv(WMMA f16 -> f32) + bias + relu + gated combine + store.
// One wave32 per (b, 16-wide s-tile). All weights (48KB f16) staged in LDS.
// D = A(16c x 32f) x B(32f x 16s): 2 K-chunks x 3 taps x 8 experts.
// ---------------------------------------------------------------------------
__global__ void __launch_bounds__(256) moe_main(
    const _Float16* __restrict__ xT, const _Float16* __restrict__ w16,
    const float* __restrict__ conv_b, const float* __restrict__ expert_bias,
    const float* __restrict__ gates, const float* __restrict__ task_bias,
    float* __restrict__ out) {
    __shared__ _Float16 wlds[E_ * 3 * C_ * F_];   // 48 KB

    for (int i = threadIdx.x; i < (E_ * 3 * C_ * F_) / 8; i += 256)
        *(v8h*)(wlds + i * 8) = *(const v8h*)(w16 + i * 8);
    __syncthreads();

    const int wave = threadIdx.x >> 5;
    const int lane = threadIdx.x & 31;
    const int n    = lane & 15;     // s-column within tile / c-row of A
    const int hi   = lane >> 4;     // lane half selects K / M halves
    const int tile = blockIdx.x * 8 + wave;       // 1600 blocks * 8 waves = 12800
    const int b    = tile / ST_;
    const int s0   = (tile % ST_) * 16;

    // B-fragments of x: [tap k][K-chunk]; lane = s-column, 16 consecutive f halves.
    // Lanes 0-15 carry K=0..15, lanes 16-31 carry K=16..31 of each 32-K chunk.
    v16h bf[3][2];
#pragma unroll
    for (int k = 0; k < 3; ++k) {
        const int s = s0 + n + k - 1;                       // SAME padding
        const bool valid = (s >= 0) && (s < S_);
        const _Float16* base =
            xT + ((size_t)b * S_ + (valid ? s : 0)) * F_ + hi * 16;
#pragma unroll
        for (int ch = 0; ch < 2; ++ch) {
            v16h v = {};
            if (valid) v = *(const v16h*)(base + ch * 32);
            bf[k][ch] = v;
        }
    }

    float oacc[T_][8];
#pragma unroll
    for (int t = 0; t < T_; ++t)
#pragma unroll
        for (int i = 0; i < 8; ++i) oacc[t][i] = 0.0f;

#pragma unroll
    for (int e = 0; e < E_; ++e) {
        v8f acc = {};
#pragma unroll
        for (int k = 0; k < 3; ++k) {
#pragma unroll
            for (int ch = 0; ch < 2; ++ch) {
                // A-fragment: row c = n; halves {0..7}->K fb+hi*16+j, {8..15}->K fb+hi*16+8+j
                const _Float16* wbase =
                    wlds + (((e * 3 + k) * C_) + n) * F_ + ch * 32 + hi * 16;
                v16h af;
                *(v8h*)&af         = *(const v8h*)(wbase);
                *(((v8h*)&af) + 1) = *(const v8h*)(wbase + 8);
                acc = __builtin_amdgcn_wmma_f32_16x16x32_f16(
                    false, af, false, bf[k][ch], (short)0, acc, false, false);
            }
        }
        float gl[T_];
#pragma unroll
        for (int t = 0; t < T_; ++t)
            gl[t] = gates[((size_t)t * B_ + b) * E_ + e];
#pragma unroll
        for (int i = 0; i < 8; ++i) {
            const int c = i + 8 * hi;                      // D: VGPR i, lane half
            const int s = s0 + n;
            float v = acc[i] + conv_b[e * C_ + c] +
                      expert_bias[(size_t)e * (C_ * S_) + c * S_ + s];
            v = fmaxf(v, 0.0f);                            // relu
#pragma unroll
            for (int t = 0; t < T_; ++t)
                oacc[t][i] = fmaf(gl[t], v, oacc[t][i]);
        }
    }

#pragma unroll
    for (int t = 0; t < T_; ++t) {
        const float tb = task_bias[t];
#pragma unroll
        for (int i = 0; i < 8; ++i) {
            const int c = i + 8 * hi;
            const int s = s0 + n;
            out[(((size_t)t * B_ + b) * C_ + c) * S_ + s] = oacc[t][i] + tb;
        }
    }
}

// ---------------------------------------------------------------------------
extern "C" void kernel_launch(void* const* d_in, const int* in_sizes, int n_in,
                              void* d_out, int out_size, void* d_ws, size_t ws_size,
                              hipStream_t stream) {
    (void)in_sizes; (void)n_in; (void)out_size; (void)ws_size;
    const float* x           = (const float*)d_in[0];
    const float* conv_w     = (const float*)d_in[1];
    const float* conv_b     = (const float*)d_in[2];
    const float* expert_bias = (const float*)d_in[3];
    const float* gk          = (const float*)d_in[4];
    const float* gate_bias   = (const float*)d_in[5];
    const float* task_bias   = (const float*)d_in[6];
    float* out = (float*)d_out;

    char* ws = (char*)d_ws;
    _Float16* xT  = (_Float16*)ws;                              // 26,214,400 B
    _Float16* w16 = (_Float16*)(ws + 26214400);                 //     49,152 B
    float*  gates = (float*)(ws + 26214400 + 49152);            //     65,536 B

    prep_x<<<(B_ * S_) / 256, 256, 0, stream>>>(x, xT);
    prep_w<<<96, 256, 0, stream>>>(conv_w, w16);
    gates_kernel<<<B_, 256, 0, stream>>>(x, gk, gate_bias, gates);
    moe_main<<<NTILE_ / 8, 256, 0, stream>>>(xT, w16, conv_b, expert_bias,
                                             gates, task_bias, out);
}
